// lstm_decoder_61323543052439
// MI455X (gfx1250) — matrix operand order repaired and checked by reading, not compiled, not run on hardware
//
#include <hip/hip_runtime.h>

// ---------------------------------------------------------------------------
// LSTM decoder for MI455X (gfx1250, wave32, WMMA).
// B=16384 batch rows are fully independent through the recurrence, so each
// 512-thread workgroup owns M=32 rows and runs all T=128 steps locally.
// Dominant GEMM per step: h[32,1024] @ W_hh^T[1024,4096] via
// v_wmma_f32_16x16x32_bf16. Weights pre-packed into WMMA B-fragment layout.
// h state: double-buffered bf16 in LDS. c state: fp32 in workspace (L2-hot).
// ---------------------------------------------------------------------------

#define BB   16384
#define TT   128
#define DD   1024
#define HH   2048
#define MM   32            // batch rows per workgroup
#define HS   1040          // LDS row stride in bf16 elems (1024 + 16 pad, 32B-mult)

typedef __attribute__((ext_vector_type(16))) __bf16 bf16x16;
typedef __attribute__((ext_vector_type(8)))  float  f32x8;
typedef __attribute__((ext_vector_type(4)))  float  f32x4;
typedef __attribute__((ext_vector_type(4)))  int    i32x4;

union FragU { i32x4 q[2]; bf16x16 v; };

// B fragment: packed buffer, fragment = 32 lanes * 32 bytes = 1024 B.
static __device__ inline bf16x16 load_frag_b(const __bf16* pack, int frag, int lane) {
    const i32x4* p = (const i32x4*)(pack + ((size_t)frag << 9)) + (lane << 1);
    FragU u; u.q[0] = p[0]; u.q[1] = p[1]; return u.v;
}

// A fragment from row-major bf16 LDS tile (stride HS).
// lanes 0-15: K 0-7 & 16-23 of row lane; lanes 16-31: K 8-15 & 24-31.
static __device__ inline bf16x16 load_frag_a(const __bf16* hbase, int rg, int kt, int lane) {
    const int row = (rg << 4) + (lane & 15);
    const int kb  = (kt << 5) + ((lane >> 4) << 3);
    const i32x4* p = (const i32x4*)(hbase + (size_t)row * HS + kb);
    FragU u; u.q[0] = p[0]; u.q[1] = p[2]; return u.v;
}

static __device__ inline float sigmf(float x) {
    return __builtin_amdgcn_rcpf(1.f + __expf(-x));
}
static __device__ inline float tanh_f(float x) {
    // 1 - 2/(e^{2x}+1): saturates correctly at +/-inf
    return 1.f - 2.f * __builtin_amdgcn_rcpf(__expf(2.f * x) + 1.f);
}

// --- pack W (fp32 [N][K] row-major, i.e. B^T) into bf16 WMMA B-fragment layout
__global__ void pack_b_kernel(const float* __restrict__ W, unsigned short* dst_u,
                              int N, int K) {
    __bf16* dst = reinterpret_cast<__bf16*>(dst_u);
    const int KC = K >> 4;             // 16-elem chunks along K
    const int tid = blockIdx.x * blockDim.x + threadIdx.x;
    if (tid >= N * KC) return;
    const int n = tid / KC, kc = tid - n * KC;
    const int nt = n >> 4, kt = kc >> 1;
    const int lane = (n & 15) | ((kc & 1) << 4);
    const int KT = K >> 5;
    const size_t off = (((size_t)(nt * KT + kt) << 5) + lane) << 4;  // *16 bf16
    const float* src = W + (size_t)n * K + ((size_t)kt << 5) + ((kc & 1) << 3);
    #pragma unroll
    for (int i = 0; i < 16; ++i) dst[off + i] = (__bf16)src[i + (i & 8)];
}

__global__ void bias_kernel(const float* __restrict__ b_ih,
                            const float* __restrict__ b_hh,
                            float* __restrict__ bias) {
    const int i = blockIdx.x * blockDim.x + threadIdx.x;
    if (i < 4 * DD) bias[i] = b_ih[i] + b_hh[i];
}

// --- h0 = h_in @ W_fch^T + b_fch  -> cws (serves as both c0 and h0 source)
__global__ __launch_bounds__(256) void gemm_h0_kernel(
        const float* __restrict__ h_in, const float* __restrict__ b_fch,
        const unsigned short* __restrict__ wpack_u, float* __restrict__ cws) {
    const __bf16* wpack = reinterpret_cast<const __bf16*>(wpack_u);
    const int lane = threadIdx.x & 31, wv = threadIdx.x >> 5;
    const int gid = blockIdx.x * 8 + wv;      // 65536 16x16 tiles
    const int bm = gid >> 6, bn = gid & 63;
    f32x8 acc = {0.f, 0.f, 0.f, 0.f, 0.f, 0.f, 0.f, 0.f};
    const int row = (bm << 4) + (lane & 15);
    const float* pa_base = h_in + (size_t)row * HH;
    for (int kt = 0; kt < (HH >> 5); ++kt) {
        const int kb = (kt << 5) + ((lane >> 4) << 3);
        const float* pa = pa_base + kb;
        f32x4 a0 = *(const f32x4*)pa;
        f32x4 a1 = *(const f32x4*)(pa + 4);
        f32x4 a2 = *(const f32x4*)(pa + 16);
        f32x4 a3 = *(const f32x4*)(pa + 20);
        bf16x16 va;
        #pragma unroll
        for (int i = 0; i < 4; ++i) {
            va[i]      = (__bf16)a0[i];
            va[4 + i]  = (__bf16)a1[i];
            va[8 + i]  = (__bf16)a2[i];
            va[12 + i] = (__bf16)a3[i];
        }
        bf16x16 vb = load_frag_b(wpack, bn * (HH >> 5) + kt, lane);
        acc = __builtin_amdgcn_wmma_f32_16x16x32_bf16(false, va, false, vb,
                                                      (short)0, acc, false, false);
    }
    const int n = (bn << 4) + (lane & 15);
    const float bb = b_fch[n];
    #pragma unroll
    for (int v = 0; v < 8; ++v) {
        const int m = (bm << 4) + v + ((lane >> 4) << 3);
        cws[(size_t)m * DD + n] = acc[v] + bb;
    }
}

// --- persistent per-batch-tile LSTM loop over all T steps
__global__ __launch_bounds__(512) void lstm_main_kernel(
        const float* __restrict__ ct, const float* __restrict__ z,
        const float* __restrict__ W_ih, const float* __restrict__ W_out,
        const float* __restrict__ b_out,
        const unsigned short* __restrict__ whh_u,
        const float* __restrict__ bias,
        float* __restrict__ cws, float* __restrict__ out) {
    const __bf16* whh = reinterpret_cast<const __bf16*>(whh_u);
    __shared__ __bf16 hbuf[2][MM * HS];      // double-buffered h (bf16)
    __shared__ float  xs[MM];                // decoder input per row
    __shared__ float  ypart[512];
    __shared__ float  wout_s[DD];

    const int tid  = threadIdx.x;
    const int lane = tid & 31, wv = tid >> 5;   // 16 waves
    const int b0   = blockIdx.x * MM;

    // init: h0/c0 from cws, x0 = ct + z[:,0], cache W_out
    for (int i = tid; i < MM * DD; i += 512) {
        const int r = i >> 10, d = i & (DD - 1);
        hbuf[0][r * HS + d] = (__bf16)cws[(size_t)(b0 + r) * DD + d];
    }
    for (int i = tid; i < DD; i += 512) wout_s[i] = W_out[i];
    if (tid < MM) xs[tid] = ct[b0 + tid] + z[(size_t)(b0 + tid) * TT];
    __syncthreads();
    const float bout = b_out[0];

    for (int t = 0; t < TT; ++t) {
        const __bf16* hcur = hbuf[t & 1];
        __bf16*       hnxt = hbuf[(t + 1) & 1];

        // wave wv owns state column groups cg = 4*wv .. 4*wv+3 (16 cols each)
        for (int cgi = 0; cgi < 4; ++cgi) {
            const int cg = (wv << 2) + cgi;
            f32x8 acc[2][4];
            #pragma unroll
            for (int rr = 0; rr < 2; ++rr)
                #pragma unroll
                for (int g = 0; g < 4; ++g)
                    acc[rr][g] = (f32x8){0.f,0.f,0.f,0.f,0.f,0.f,0.f,0.f};

            for (int kt = 0; kt < (DD >> 5); ++kt) {
                const bf16x16 a0 = load_frag_a(hcur, 0, kt, lane);
                const bf16x16 a1 = load_frag_a(hcur, 1, kt, lane);
                #pragma unroll
                for (int g = 0; g < 4; ++g) {
                    const int frag = (((g << 6) + cg) << 5) + kt;  // nt*KT + kt
                    const bf16x16 b = load_frag_b(whh, frag, lane);
                    acc[0][g] = __builtin_amdgcn_wmma_f32_16x16x32_bf16(
                        false, a0, false, b, (short)0, acc[0][g], false, false);
                    acc[1][g] = __builtin_amdgcn_wmma_f32_16x16x32_bf16(
                        false, a1, false, b, (short)0, acc[1][g], false, false);
                }
            }

            // fused epilogue: bias + x*W_ih + activations + c/h update
            const int n = (cg << 4) + (lane & 15);
            const float bi = bias[n],          bf_ = bias[DD + n];
            const float bg = bias[2 * DD + n], bo  = bias[3 * DD + n];
            const float wi = W_ih[n],          wf  = W_ih[DD + n];
            const float wg = W_ih[2 * DD + n], wo  = W_ih[3 * DD + n];
            #pragma unroll
            for (int rr = 0; rr < 2; ++rr) {
                #pragma unroll
                for (int v = 0; v < 8; ++v) {
                    const int m = (rr << 4) + v + ((lane >> 4) << 3);
                    const float x = xs[m];
                    const float gi = acc[rr][0][v] + bi + x * wi;
                    const float gf = acc[rr][1][v] + bf_ + x * wf;
                    const float gg = acc[rr][2][v] + bg + x * wg;
                    const float go = acc[rr][3][v] + bo + x * wo;
                    const size_t ci = (size_t)(b0 + m) * DD + n;
                    const float c = sigmf(gf) * cws[ci] + sigmf(gi) * tanh_f(gg);
                    cws[ci] = c;
                    hnxt[m * HS + n] = (__bf16)(sigmf(go) * tanh_f(c));
                }
            }
        }
        __syncthreads();   // all of h_t written

        // y_t = h_t @ W_out^T + b_out ; x_{t+1} = y_t + z[:,t+1]
        {
            const int r = tid >> 4, p = tid & 15;
            const __bf16* hr = hnxt + r * HS + (p << 6);
            const float*  wr = wout_s + (p << 6);
            float s = 0.f;
            for (int dd = 0; dd < 64; ++dd) s += (float)hr[dd] * wr[dd];
            ypart[tid] = s;
        }
        __syncthreads();
        if (tid < MM) {
            float y = bout;
            #pragma unroll
            for (int j = 0; j < 16; ++j) y += ypart[(tid << 4) + j];
            out[(size_t)(b0 + tid) * TT + t] = y;
            xs[tid] = (t + 1 < TT) ? (y + z[(size_t)(b0 + tid) * TT + t + 1]) : 0.f;
        }
        __syncthreads();
    }
}

// ---------------------------------------------------------------------------
// workspace layout
static constexpr size_t SZ_WHH  = (size_t)4 * DD * DD * 2;   //  8 MB bf16 pack
static constexpr size_t SZ_WFCH = (size_t)DD * HH * 2;       //  4 MB bf16 pack
static constexpr size_t OFF_WHH  = 0;
static constexpr size_t OFF_WFCH = OFF_WHH + SZ_WHH;
static constexpr size_t OFF_BIAS = OFF_WFCH + SZ_WFCH;
static constexpr size_t OFF_C    = OFF_BIAS + 65536;         // c state: 64 MB fp32

extern "C" void kernel_launch(void* const* d_in, const int* in_sizes, int n_in,
                              void* d_out, int out_size, void* d_ws, size_t ws_size,
                              hipStream_t stream) {
    (void)in_sizes; (void)n_in; (void)out_size; (void)ws_size;
    const float* ct    = (const float*)d_in[0];
    const float* z     = (const float*)d_in[1];
    const float* h_in  = (const float*)d_in[2];
    const float* W_fch = (const float*)d_in[3];
    const float* b_fch = (const float*)d_in[4];
    const float* W_ih  = (const float*)d_in[5];
    const float* W_hh  = (const float*)d_in[6];
    const float* b_ih  = (const float*)d_in[7];
    const float* b_hh  = (const float*)d_in[8];
    const float* W_out = (const float*)d_in[9];
    const float* b_out = (const float*)d_in[10];

    char* ws = (char*)d_ws;
    unsigned short* whh_p  = (unsigned short*)(ws + OFF_WHH);
    unsigned short* wfch_p = (unsigned short*)(ws + OFF_WFCH);
    float*          bias   = (float*)(ws + OFF_BIAS);
    float*          cws    = (float*)(ws + OFF_C);
    float*          out    = (float*)d_out;

    // 1) pack weights into WMMA B-fragment layout (bf16)
    pack_b_kernel<<<(4 * DD * (DD >> 4) + 255) / 256, 256, 0, stream>>>(W_hh, whh_p, 4 * DD, DD);
    pack_b_kernel<<<(DD * (HH >> 4) + 255) / 256, 256, 0, stream>>>(W_fch, wfch_p, DD, HH);
    bias_kernel<<<(4 * DD + 255) / 256, 256, 0, stream>>>(b_ih, b_hh, bias);

    // 2) h0 = c0 = h_in @ W_fch^T + b_fch   (WMMA GEMM into cws)
    gemm_h0_kernel<<<(BB / 16) * (DD / 16) / 8, 256, 0, stream>>>(h_in, b_fch, wfch_p, cws);

    // 3) persistent per-batch-tile LSTM over all T steps (WMMA per step)
    lstm_main_kernel<<<BB / MM, 512, 0, stream>>>(ct, z, W_ih, W_out, b_out,
                                                  whh_p, bias, cws, out);
}
